// HexGNN_4698694222466
// MI455X (gfx1250) — matrix operand order repaired and checked by reading, not compiled, not run on hardware
//
#include <hip/hip_runtime.h>

#define NN 169
#define NPAD 176
#define HDIM 128
#define NLAYERS 8
#define NCH 8

typedef __bf16 bf16_t;
typedef bf16_t v16bf __attribute__((ext_vector_type(16)));
typedef float v8f __attribute__((ext_vector_type(8)));

// ---------- helpers ----------
__device__ __forceinline__ unsigned short f2bf(float f) {
  unsigned int u = __builtin_bit_cast(unsigned int, f);
  u += 0x7FFFu + ((u >> 16) & 1u);          // round-to-nearest-even
  return (unsigned short)(u >> 16);
}
__device__ __forceinline__ float bf2f(unsigned short h) {
  unsigned int u = ((unsigned int)h) << 16;
  return __builtin_bit_cast(float, u);
}
__device__ __forceinline__ float gelu_exact(float x) {
  return 0.5f * x * (1.0f + erff(x * 0.7071067811865475f));
}

union V16U { uint4 q[2]; v16bf v; };

// A operand (16x32 bf16 tile) from row-major LDS [NPAD][HDIM].
// Lane L: row M = mt*16 + (L&15); khalf = L>>4 selects K chunks
// {khalf*8 .. +7} and {16+khalf*8 .. +7} per CDNA5 ISA 7.12.2.
__device__ __forceinline__ v16bf load_A_lds(const unsigned short* base,
                                            int row, int kbase, int khalf) {
  V16U u;
  const unsigned short* rp = base + row * HDIM + kbase + khalf * 8;
  u.q[0] = *(const uint4*)(rp);
  u.q[1] = *(const uint4*)(rp + 16);
  return u.v;
}

// B operand (32x16 bf16 tile) from pre-transposed global weights Wt[out][K].
// Lane L: column N = nt*16 + (L&15); K chunks kt*32 + {(L>>4)*8 .. +7} and {16+(L>>4)*8 .. +7}.
__device__ __forceinline__ v16bf load_B_glb(const unsigned short* wt, int K,
                                            int ncol, int kt, int khalf) {
  V16U u;
  const unsigned short* p = wt + ncol * K + kt * 32 + khalf * 8;
  u.q[0] = *(const uint4*)(p);
  u.q[1] = *(const uint4*)(p + 16);
  return u.v;
}

__device__ __forceinline__ v8f wmma_bf16(v16bf a, v16bf b, v8f c) {
  return __builtin_amdgcn_wmma_f32_16x16x32_bf16(false, a, false, b,
                                                 (short)0, c, false, false);
}

// ---------- weight transpose/convert kernel (fp32 [K][N] -> bf16 [N][K]) ----------
__global__ void hexgnn_convert_weights(const float* __restrict__ msg_w,
                                       const float* __restrict__ upd_w,
                                       unsigned short* __restrict__ msg_wt,
                                       unsigned short* __restrict__ upd_wt) {
  int idx = blockIdx.x * blockDim.x + threadIdx.x;
  if (idx < NLAYERS * HDIM * HDIM) {           // msg: [l][out=128][in=128]
    int l = idx >> 14, rem = idx & 16383;
    int o = rem >> 7, k = rem & 127;
    msg_wt[idx] = f2bf(msg_w[(l << 14) + k * HDIM + o]);
  }
  if (idx < NLAYERS * HDIM * 2 * HDIM) {       // upd: [l][out=128][in=256]
    int l = idx >> 15, rem = idx & 32767;
    int o = rem >> 8, k = rem & 255;
    upd_wt[idx] = f2bf(upd_w[(l << 15) + k * HDIM + o]);
  }
}

// ---------- LDS layout (bytes) ----------
#define H32_OFF 0
#define H16_OFF (NPAD * HDIM * 4)                       // 90112
#define MSG_OFF (H16_OFF + NPAD * HDIM * 2)             // 135168
#define AGG_OFF (MSG_OFF + NPAD * HDIM * 2)             // 180224
#define SMEM_BYTES (AGG_OFF + NPAD * HDIM * 2)          // 225280

// ---------- main kernel: one block per batch element ----------
__global__ __launch_bounds__(256) void hexgnn_forward(
    const float* __restrict__ x,
    const float* __restrict__ in_w, const float* __restrict__ in_b,
    const float* __restrict__ in_g, const float* __restrict__ in_beta,
    const unsigned short* __restrict__ msg_wt,
    const unsigned short* __restrict__ upd_wt,
    const float* __restrict__ upd_b,
    const float* __restrict__ ln_g, const float* __restrict__ ln_b,
    const float* __restrict__ aq_w, const float* __restrict__ aq_b,
    const float* __restrict__ pp_w, const float* __restrict__ pp_b,
    const float* __restrict__ v1_w, const float* __restrict__ v1_b,
    const float* __restrict__ v2_w, const float* __restrict__ v2_b,
    const float* __restrict__ tf_w, const float* __restrict__ tf_b,
    float* __restrict__ out_p, float* __restrict__ out_v,
    float* __restrict__ out_t) {
  extern __shared__ char smem[];
  float*          h32   = (float*)(smem + H32_OFF);
  unsigned short* h16   = (unsigned short*)(smem + H16_OFF);
  unsigned short* msg16 = (unsigned short*)(smem + MSG_OFF);
  unsigned short* agg16 = (unsigned short*)(smem + AGG_OFF);

  const int tid  = threadIdx.x;
  const int b    = blockIdx.x;
  const int wave = tid >> 5;
  const int lane = tid & 31;
  const int khalf = lane >> 4;            // K-half selector
  const int arow  = lane & 15;            // A row within M tile
  const int ncol  = (wave << 4) + (lane & 15);  // output dim 0..127

  // zero padded rows of bf16 buffers used as WMMA A operands
  for (int idx = tid; idx < (NPAD - NN) * HDIM; idx += 256) {
    int row = NN + (idx >> 7), d = idx & 127;
    h16[row * HDIM + d] = 0;
    agg16[row * HDIM + d] = 0;
  }

  // stage x (CH x NN) into LDS (aliases msg buffer, unused yet)
  float* xs = (float*)(smem + MSG_OFF);
  for (int idx = tid; idx < NCH * NN; idx += 256)
    xs[idx] = x[b * NCH * NN + idx];
  __syncthreads();

  // input projection: h = gelu(node_feat @ in_w + in_b)
  for (int idx = tid; idx < NN * HDIM; idx += 256) {
    int n = idx >> 7, d = idx & 127;
    float acc = in_b[d];
#pragma unroll
    for (int c = 0; c < NCH; c++) acc += xs[c * NN + n] * in_w[c * HDIM + d];
    h32[n * HDIM + d] = gelu_exact(acc);
  }
  __syncthreads();

  // LayerNorm lambda (per-node, one thread per node), writes f32 + bf16
  auto layer_norm = [&](const float* g, const float* bt) {
    if (tid < NN) {
      float* row = h32 + tid * HDIM;
      unsigned short* r16 = h16 + tid * HDIM;
      float mu = 0.f;
      for (int k = 0; k < HDIM; k++) mu += row[k];
      mu *= (1.f / HDIM);
      float var = 0.f;
      for (int k = 0; k < HDIM; k++) { float d = row[k] - mu; var += d * d; }
      var *= (1.f / HDIM);
      float rs = rsqrtf(var + 1e-5f);
      for (int k = 0; k < HDIM; k++) {
        float v = (row[k] - mu) * rs * g[k] + bt[k];
        row[k] = v;
        r16[k] = f2bf(v);
      }
    }
    __syncthreads();
  };
  layer_norm(in_g, in_beta);

  // ---------------- GNN layers ----------------
  for (int l = 0; l < NLAYERS; ++l) {
    // ---- msg = h @ msg_w[l]  (169x128 @ 128x128) ----
    const unsigned short* mw = msg_wt + l * (HDIM * HDIM);
    v16bf bm[4];
#pragma unroll
    for (int kt = 0; kt < 4; kt++) bm[kt] = load_B_glb(mw, HDIM, ncol, kt, khalf);
    for (int mt = 0; mt < 11; ++mt) {
      v8f c = {0.f, 0.f, 0.f, 0.f, 0.f, 0.f, 0.f, 0.f};
#pragma unroll
      for (int kt = 0; kt < 4; kt++) {
        v16bf a = load_A_lds(h16, mt * 16 + arow, kt * 32, khalf);
        c = wmma_bf16(a, bm[kt], c);
      }
#pragma unroll
      for (int r = 0; r < 8; r++) {
        int node = mt * 16 + khalf * 8 + r;
        if (node < NN) msg16[node * HDIM + ncol] = f2bf(c[r]);
      }
    }
    __syncthreads();

    // ---- agg[v] = mean over hex neighbors of msg[u] ----
    for (int idx = tid; idx < NN * HDIM; idx += 256) {
      int node = idx >> 7, d = idx & 127;
      int i = node / 13, j = node - i * 13;
      const int di[6] = {1, -1, 0, 0, 1, -1};
      const int dj[6] = {0, 0, 1, -1, -1, 1};
      float s = 0.f; int cnt = 0;
#pragma unroll
      for (int e = 0; e < 6; e++) {
        int ni = i + di[e], nj = j + dj[e];
        if (ni >= 0 && ni < 13 && nj >= 0 && nj < 13) {
          s += bf2f(msg16[(ni * 13 + nj) * HDIM + d]);
          cnt++;
        }
      }
      agg16[node * HDIM + d] = f2bf(s / (float)cnt);
    }
    __syncthreads();

    // ---- out = concat(h, agg) @ upd_w[l] + upd_b[l]; h += gelu(out) ----
    const unsigned short* uw = upd_wt + l * (HDIM * 2 * HDIM);
    v16bf bu[8];
#pragma unroll
    for (int kt = 0; kt < 8; kt++) bu[kt] = load_B_glb(uw, 2 * HDIM, ncol, kt, khalf);
    const float ubias = upd_b[l * HDIM + ncol];
    for (int mt = 0; mt < 11; ++mt) {
      v8f c = {0.f, 0.f, 0.f, 0.f, 0.f, 0.f, 0.f, 0.f};
#pragma unroll
      for (int kt = 0; kt < 4; kt++) {
        v16bf a = load_A_lds(h16, mt * 16 + arow, kt * 32, khalf);
        c = wmma_bf16(a, bu[kt], c);
      }
#pragma unroll
      for (int kt = 4; kt < 8; kt++) {
        v16bf a = load_A_lds(agg16, mt * 16 + arow, (kt - 4) * 32, khalf);
        c = wmma_bf16(a, bu[kt], c);
      }
#pragma unroll
      for (int r = 0; r < 8; r++) {
        int node = mt * 16 + khalf * 8 + r;
        if (node < NN) {
          float v = c[r] + ubias;
          h32[node * HDIM + ncol] = gelu_exact(v) + h32[node * HDIM + ncol];
        }
      }
    }
    __syncthreads();
    layer_norm(ln_g + l * HDIM, ln_b + l * HDIM);
  }

  // ---------------- heads ----------------
  float* sc   = (float*)(smem + MSG_OFF);   // reuse msg region as f32 scratch
  float* gvec = sc + 192;
  float* hid  = gvec + 128;
  float* red  = hid + 256;

  if (tid < NN) {
    const float* hr = h32 + tid * HDIM;
    float sa = aq_b[0], sp = pp_b[0];
    for (int k = 0; k < HDIM; k++) {
      float hv = hr[k];
      sa += hv * aq_w[k];
      sp += hv * pp_w[k];
    }
    sc[tid] = sa;
    out_p[b * NN + tid] = sp;
  }
  __syncthreads();

  // softmax over 169 (redundant per-thread reduction, race-free)
  float m = -1e30f;
  for (int k = 0; k < NN; k++) m = fmaxf(m, sc[k]);
  float ssum = 0.f;
  for (int k = 0; k < NN; k++) ssum += expf(sc[k] - m);
  float inv = 1.0f / ssum;
  __syncthreads();
  if (tid < NN) sc[tid] = expf(sc[tid] - m) * inv;
  __syncthreads();

  // g = sum_n h[n] * attn[n]
  if (tid < HDIM) {
    float s = 0.f;
    for (int n = 0; n < NN; n++) s += h32[n * HDIM + tid] * sc[n];
    gvec[tid] = s;
  }
  __syncthreads();

  // hid = gelu(g @ v1_w + v1_b)
  {
    float s = v1_b[tid];
    for (int k = 0; k < HDIM; k++) s += gvec[k] * v1_w[k * 256 + tid];
    hid[tid] = gelu_exact(s);
  }
  __syncthreads();

  // v = tanh(hid @ v2_w + v2_b)
  red[tid] = hid[tid] * v2_w[tid];
  __syncthreads();
  for (int st = 128; st > 0; st >>= 1) {
    if (tid < st) red[tid] += red[tid + st];
    __syncthreads();
  }
  if (tid == 0) out_v[b] = tanhf(red[0] + v2_b[0]);

  // t = g @ tf_w + tf_b
  if (tid < 4) {
    float s = tf_b[tid];
    for (int k = 0; k < HDIM; k++) s += gvec[k] * tf_w[k * 4 + tid];
    out_t[b * 4 + tid] = s;
  }
}

// ---------------- launch ----------------
extern "C" void kernel_launch(void* const* d_in, const int* in_sizes, int n_in,
                              void* d_out, int out_size, void* d_ws, size_t ws_size,
                              hipStream_t stream) {
  const float* x       = (const float*)d_in[0];
  // d_in[1]=src, d_in[2]=dst (int32) — adjacency recomputed from geometry
  const float* in_w    = (const float*)d_in[3];
  const float* in_b    = (const float*)d_in[4];
  const float* in_g    = (const float*)d_in[5];
  const float* in_beta = (const float*)d_in[6];
  const float* msg_w   = (const float*)d_in[7];
  const float* upd_w   = (const float*)d_in[8];
  const float* upd_b   = (const float*)d_in[9];
  const float* ln_g    = (const float*)d_in[10];
  const float* ln_b    = (const float*)d_in[11];
  const float* aq_w    = (const float*)d_in[12];
  const float* aq_b    = (const float*)d_in[13];
  const float* pp_w    = (const float*)d_in[14];
  const float* pp_b    = (const float*)d_in[15];
  const float* v1_w    = (const float*)d_in[16];
  const float* v1_b    = (const float*)d_in[17];
  const float* v2_w    = (const float*)d_in[18];
  const float* v2_b    = (const float*)d_in[19];
  const float* tf_w    = (const float*)d_in[20];
  const float* tf_b    = (const float*)d_in[21];

  const int B = in_sizes[0] / (NCH * NN);

  unsigned short* msg_wt = (unsigned short*)d_ws;                     // 8*128*128
  unsigned short* upd_wt = msg_wt + NLAYERS * HDIM * HDIM;            // 8*128*256

  float* out_p = (float*)d_out;            // (B, 169)
  float* out_v = out_p + B * NN;           // (B, 1)
  float* out_t = out_v + B;                // (B, 4)

  // 1) transpose + bf16-convert weights into workspace
  {
    int total = NLAYERS * HDIM * 2 * HDIM;                 // 262144 (covers msg too)
    hexgnn_convert_weights<<<(total + 255) / 256, 256, 0, stream>>>(
        msg_w, upd_w, msg_wt, upd_wt);
  }

  // 2) main fused forward: one workgroup per batch element, all-LDS resident
  hexgnn_forward<<<B, 256, SMEM_BYTES, stream>>>(
      x, in_w, in_b, in_g, in_beta, msg_wt, upd_wt, upd_b, ln_g, ln_b,
      aq_w, aq_b, pp_w, pp_b, v1_w, v1_b, v2_w, v2_b, tf_w, tf_b,
      out_p, out_v, out_t);
}
